// Hawk_58351425683810
// MI455X (gfx1250) — matrix operand order repaired and checked by reading, not compiled, not run on hardware
//
#include <hip/hip_runtime.h>
#include <hip/hip_bf16.h>
#include <math.h>

// ---------------- problem constants ----------------
#define DIMc   1024
#define HIDc   1536
#define KCONV  4
#define NBAT   4
#define TLEN   4096
#define MROWS  (NBAT * TLEN)   // 16384 rows for all GEMMs
#define SEGS   32
#define SEGL   (TLEN / SEGS)   // 128

// ---------------- CDNA5 WMMA types ----------------
typedef __attribute__((ext_vector_type(16))) __bf16        v16bf;
typedef __attribute__((ext_vector_type(8)))  float         v8f;
typedef __attribute__((ext_vector_type(4)))  unsigned int  u32x4;

union FragU { u32x4 q[2]; v16bf v; };

// ---------------- bf16 helpers (RNE) ----------------
__device__ __forceinline__ unsigned short f2bf(float f) {
    unsigned u = __float_as_uint(f);
    u += 0x7fffu + ((u >> 16) & 1u);
    return (unsigned short)(u >> 16);
}
__device__ __forceinline__ float bf2f(unsigned short s) {
    return __uint_as_float(((unsigned)s) << 16);
}
__device__ __forceinline__ float sigm(float x) { return 1.0f / (1.0f + expf(-x)); }
__device__ __forceinline__ float gelu_exact(float x) {
    return 0.5f * x * (1.0f + erff(x * 0.70710678118654752f));
}

// ---------------- fp32 -> bf16 convert ----------------
__global__ void f32_to_bf16(const float* __restrict__ in,
                            unsigned short* __restrict__ out, size_t n) {
    size_t i = (size_t)blockIdx.x * blockDim.x + threadIdx.x;
    if (i < n) out[i] = f2bf(in[i]);
}

// ---------------- WMMA GEMM: C(M,Nc) = A(M,K) * W(Nc,K)^T ----------------
// Wave tile 64x64 (4x4 WMMA subtiles), block tile 128x256 (8 waves, 2x4).
// MODE 0: bf16 split store (col<split -> ob0, else ob1), leading dim ldo
// MODE 1: bf16 store to ob0, leading dim ldo
// MODE 2: fp32 store to of, leading dim ldo
template <int MODE>
__global__ __launch_bounds__(256) void gemm_wmma_bf16(
    const unsigned short* __restrict__ A,
    const unsigned short* __restrict__ W,
    int Kd,
    unsigned short* __restrict__ ob0,
    unsigned short* __restrict__ ob1,
    float* __restrict__ of,
    int ldo, int split) {
    const int lane = threadIdx.x & 31;
    const int wave = threadIdx.x >> 5;
    const int wm   = wave & 1;     // 2 waves along M
    const int wn   = wave >> 1;    // 4 waves along N
    const int m0   = blockIdx.y * 128 + wm * 64;
    const int n0   = blockIdx.x * 256 + wn * 64;
    const int lg   = lane >> 4;    // half-wave group
    const int lr   = lane & 15;

    v8f acc[4][4] = {};

    for (int k0 = 0; k0 < Kd; k0 += 32) {
        FragU au[4], bu[4];
        // A fragments: lane holds row (m0+16i+lr), k = k0 + lg*8 + {0..7} and +16
#pragma unroll
        for (int i = 0; i < 4; ++i) {
            const unsigned short* pa =
                A + (size_t)(m0 + 16 * i + lr) * Kd + k0 + lg * 8;
            au[i].q[0] = *(const u32x4*)(pa);
            au[i].q[1] = *(const u32x4*)(pa + 16);
            if (i == 0) __builtin_prefetch(pa + 32, 0, 3);  // next K-chunk
        }
        // B fragments: lane holds W row (n0+16j+lr), k = k0 + lg*8 + {0..7} and +16
#pragma unroll
        for (int j = 0; j < 4; ++j) {
            const unsigned short* pb =
                W + (size_t)(n0 + 16 * j + lr) * Kd + k0 + lg * 8;
            bu[j].q[0] = *(const u32x4*)(pb);
            bu[j].q[1] = *(const u32x4*)(pb + 16);
        }
#pragma unroll
        for (int i = 0; i < 4; ++i) {
#pragma unroll
            for (int j = 0; j < 4; ++j) {
                acc[i][j] = __builtin_amdgcn_wmma_f32_16x16x32_bf16(
                    false, au[i].v, false, bu[j].v, (short)0, acc[i][j],
                    false, false);
            }
        }
    }

    // store: C VGPR r -> row m0+16i+8*lg+r, col n0+16j+lr
#pragma unroll
    for (int i = 0; i < 4; ++i) {
#pragma unroll
        for (int j = 0; j < 4; ++j) {
            const int col = n0 + 16 * j + lr;
#pragma unroll
            for (int r = 0; r < 8; ++r) {
                const int row = m0 + 16 * i + 8 * lg + r;
                const float v = acc[i][j][r];
                if (MODE == 0) {
                    if (col < split)
                        ob0[(size_t)row * ldo + col] = f2bf(v);
                    else
                        ob1[(size_t)row * ldo + (col - split)] = f2bf(v);
                } else if (MODE == 1) {
                    ob0[(size_t)row * ldo + col] = f2bf(v);
                } else {
                    of[(size_t)row * ldo + col] = v;
                }
            }
        }
    }
}

// ---------------- causal depthwise conv (K=4) ----------------
__global__ void dwconv_kernel(const unsigned short* __restrict__ xh,
                              const float* __restrict__ w,
                              const float* __restrict__ bias,
                              unsigned short* __restrict__ xc) {
    size_t idx = (size_t)blockIdx.x * blockDim.x + threadIdx.x;
    const size_t total = (size_t)MROWS * HIDc;
    if (idx >= total) return;
    const int h  = (int)(idx % HIDc);
    const size_t bt = idx / HIDc;
    const int t  = (int)(bt % TLEN);
    float acc = bias[h];
#pragma unroll
    for (int k = 0; k < KCONV; ++k) {
        const int dt = KCONV - 1 - k;  // source offset t - dt
        if (t - dt >= 0)
            acc += w[h * KCONV + k] * bf2f(xh[(bt - dt) * HIDc + h]);
    }
    xc[idx] = f2bf(acc);
}

// ---------------- gate math shared by scan phases ----------------
__device__ __forceinline__ void gate_math(const unsigned short* __restrict__ g,
                                          const unsigned short* __restrict__ xc,
                                          size_t row, int h, float sp8,
                                          float bgf, float bgi,
                                          float& alpha, float& xg) {
    const float fb = bf2f(g[row * (2 * HIDc) + h]) + bgf;
    const float ip = bf2f(g[row * (2 * HIDc) + HIDc + h]) + bgi;
    const float xv = bf2f(xc[row * HIDc + h]);
    alpha = expf(sp8 * sigm(fb));
    const float beta = sqrtf(fmaxf(1.0f - alpha * alpha + 1e-6f, 0.0f));
    xg = beta * sigm(ip) * xv;
}

// ---------------- scan phase 1: per-segment (A,B) ----------------
__global__ void scan_phase1(const unsigned short* __restrict__ g,
                            const unsigned short* __restrict__ xc,
                            const float* __restrict__ bg,
                            const float* __restrict__ fbase,
                            float* __restrict__ segA,
                            float* __restrict__ segB) {
    const int idx = blockIdx.x * blockDim.x + threadIdx.x;
    const int total = NBAT * SEGS * HIDc;
    if (idx >= total) return;
    const int h   = idx % HIDc;
    const int seg = (idx / HIDc) % SEGS;
    const int b   = idx / (HIDc * SEGS);
    const float sp8 = -8.0f * log1pf(expf(fbase[h]));
    const float bgf = bg[h], bgi = bg[HIDc + h];
    float aAcc = 1.0f, bAcc = 0.0f;
    const size_t rowbase = (size_t)b * TLEN + (size_t)seg * SEGL;
    for (int t = 0; t < SEGL; ++t) {
        float alpha, xg;
        gate_math(g, xc, rowbase + t, h, sp8, bgf, bgi, alpha, xg);
        aAcc *= alpha;
        bAcc = alpha * bAcc + xg;
    }
    segA[idx] = aAcc;
    segB[idx] = bAcc;
}

// ---------------- scan phase 2: sequential carry across segments ----------------
__global__ void scan_phase2(const float* __restrict__ segA,
                            const float* __restrict__ segB,
                            float* __restrict__ carry) {
    const int idx = blockIdx.x * blockDim.x + threadIdx.x;
    const int total = NBAT * HIDc;
    if (idx >= total) return;
    const int h = idx % HIDc;
    const int b = idx / HIDc;
    float c = 0.0f;
    for (int s = 0; s < SEGS; ++s) {
        const int o = (b * SEGS + s) * HIDc + h;
        carry[o] = c;
        c = segA[o] * c + segB[o];
    }
}

// ---------------- scan phase 3: replay with carry, fuse gelu(gate)*h ----------------
__global__ void scan_phase3(const unsigned short* __restrict__ g,
                            const unsigned short* __restrict__ xc,
                            const unsigned short* __restrict__ gate,
                            const float* __restrict__ bg,
                            const float* __restrict__ fbase,
                            const float* __restrict__ carry,
                            unsigned short* __restrict__ ah) {
    const int idx = blockIdx.x * blockDim.x + threadIdx.x;
    const int total = NBAT * SEGS * HIDc;
    if (idx >= total) return;
    const int h   = idx % HIDc;
    const int seg = (idx / HIDc) % SEGS;
    const int b   = idx / (HIDc * SEGS);
    const float sp8 = -8.0f * log1pf(expf(fbase[h]));
    const float bgf = bg[h], bgi = bg[HIDc + h];
    float hcar = carry[idx];
    const size_t rowbase = (size_t)b * TLEN + (size_t)seg * SEGL;
    for (int t = 0; t < SEGL; ++t) {
        const size_t row = rowbase + t;
        float alpha, xg;
        gate_math(g, xc, row, h, sp8, bgf, bgi, alpha, xg);
        hcar = alpha * hcar + xg;
        const float gt = bf2f(gate[row * HIDc + h]);
        ah[row * HIDc + h] = f2bf(gelu_exact(gt) * hcar);
    }
}

// ---------------- host-side launch ----------------
extern "C" void kernel_launch(void* const* d_in, const int* in_sizes, int n_in,
                              void* d_out, int out_size, void* d_ws, size_t ws_size,
                              hipStream_t stream) {
    (void)in_sizes; (void)n_in; (void)out_size; (void)ws_size;
    const float* x      = (const float*)d_in[0];
    const float* Wi     = (const float*)d_in[1];
    const float* conv_w = (const float*)d_in[2];
    const float* conv_b = (const float*)d_in[3];
    const float* Wg     = (const float*)d_in[4];
    const float* bg     = (const float*)d_in[5];
    const float* fbase  = (const float*)d_in[6];
    const float* Wo     = (const float*)d_in[7];
    float* out = (float*)d_out;

    char* p = (char*)d_ws;
    auto alloc = [&](size_t bytes) -> char* {
        char* r = p;
        p += (bytes + 255) & ~(size_t)255;
        return r;
    };
    unsigned short* xb    = (unsigned short*)alloc((size_t)MROWS * DIMc * 2);
    unsigned short* Wib   = (unsigned short*)alloc((size_t)2 * HIDc * DIMc * 2);
    unsigned short* Wgb   = (unsigned short*)alloc((size_t)2 * HIDc * HIDc * 2);
    unsigned short* Wob   = (unsigned short*)alloc((size_t)DIMc * HIDc * 2);
    unsigned short* gateb = (unsigned short*)alloc((size_t)MROWS * HIDc * 2);
    unsigned short* xhb   = (unsigned short*)alloc((size_t)MROWS * HIDc * 2);
    unsigned short* xcb   = (unsigned short*)alloc((size_t)MROWS * HIDc * 2);
    unsigned short* gb    = (unsigned short*)alloc((size_t)MROWS * 2 * HIDc * 2);
    unsigned short* ahb   = (unsigned short*)alloc((size_t)MROWS * HIDc * 2);
    float* segA  = (float*)alloc((size_t)NBAT * SEGS * HIDc * 4);
    float* segB  = (float*)alloc((size_t)NBAT * SEGS * HIDc * 4);
    float* carry = (float*)alloc((size_t)NBAT * SEGS * HIDc * 4);

    // 1) downcast inputs to bf16
    {
        size_t n;
        n = (size_t)MROWS * DIMc;
        f32_to_bf16<<<(unsigned)((n + 255) / 256), 256, 0, stream>>>(x, xb, n);
        n = (size_t)2 * HIDc * DIMc;
        f32_to_bf16<<<(unsigned)((n + 255) / 256), 256, 0, stream>>>(Wi, Wib, n);
        n = (size_t)2 * HIDc * HIDc;
        f32_to_bf16<<<(unsigned)((n + 255) / 256), 256, 0, stream>>>(Wg, Wgb, n);
        n = (size_t)DIMc * HIDc;
        f32_to_bf16<<<(unsigned)((n + 255) / 256), 256, 0, stream>>>(Wo, Wob, n);
    }

    // 2) GEMM1: proj = x @ Wi^T, split into gate | xh
    {
        dim3 grid(2 * HIDc / 256, MROWS / 128);
        gemm_wmma_bf16<0><<<grid, 256, 0, stream>>>(xb, Wib, DIMc, gateb, xhb,
                                                    nullptr, HIDc, HIDc);
    }

    // 3) causal depthwise conv
    {
        size_t n = (size_t)MROWS * HIDc;
        dwconv_kernel<<<(unsigned)((n + 255) / 256), 256, 0, stream>>>(
            xhb, conv_w, conv_b, xcb);
    }

    // 4) GEMM2: g = xc @ Wg^T
    {
        dim3 grid(2 * HIDc / 256, MROWS / 128);
        gemm_wmma_bf16<1><<<grid, 256, 0, stream>>>(xcb, Wgb, HIDc, gb, nullptr,
                                                    nullptr, 2 * HIDc, 0);
    }

    // 5) chunked RG-LRU scan (3 phases), fusing gate math + gelu(gate)*h
    {
        const int t1 = NBAT * SEGS * HIDc;
        scan_phase1<<<(t1 + 255) / 256, 256, 0, stream>>>(gb, xcb, bg, fbase,
                                                          segA, segB);
        const int t2 = NBAT * HIDc;
        scan_phase2<<<(t2 + 255) / 256, 256, 0, stream>>>(segA, segB, carry);
        scan_phase3<<<(t1 + 255) / 256, 256, 0, stream>>>(gb, xcb, gateb, bg,
                                                          fbase, carry, ahb);
    }

    // 6) GEMM3: out = ah @ Wo^T (fp32 store)
    {
        dim3 grid(DIMc / 256, MROWS / 128);
        gemm_wmma_bf16<2><<<grid, 256, 0, stream>>>(ahb, Wob, HIDc, nullptr,
                                                    nullptr, out, DIMc, 0);
    }
}
